// ResBlock_20134806684021
// MI455X (gfx1250) — matrix operand order repaired and checked by reading, not compiled, not run on hardware
//
#include <hip/hip_runtime.h>
#include <stdint.h>

#define NPTS 400000
#define CCH  32
#define KK   27
#define EPSV 1e-5f
#define NTILES (NPTS / 16)   // 25000, exact

typedef __attribute__((ext_vector_type(16))) __bf16        v16bf;
typedef __attribute__((ext_vector_type(8)))  float         v8f;
typedef __attribute__((ext_vector_type(4)))  unsigned int  v4u;

union ABu { v16bf v; v4u u[2]; };

__device__ __forceinline__ unsigned short f2bf(float f) {
    union { float f; unsigned u; } x; x.f = f;
    unsigned r = x.u + 0x7FFFu + ((x.u >> 16) & 1u);   // round-to-nearest-even
    return (unsigned short)(r >> 16);
}

// ---- elementwise f32 -> bf16 ----
__global__ void cvt_bf16(const float* __restrict__ in, unsigned short* __restrict__ out, int n) {
    for (int i = blockIdx.x * blockDim.x + threadIdx.x; i < n; i += gridDim.x * blockDim.x)
        out[i] = f2bf(in[i]);
}

// ---- swizzle weights [K][Cin][Cout] f32 -> B-operand layout [k][t][lane][16] bf16 ----
__global__ void cvt_w(const float* __restrict__ W, unsigned short* __restrict__ WB) {
    int tid = blockIdx.x * blockDim.x + threadIdx.x;
    if (tid >= KK * 1024) return;
    int j    = tid & 15;
    int lane = (tid >> 4) & 31;
    int t    = (tid >> 9) & 1;
    int k    = tid >> 10;
    int krow = (lane >> 4) * 16 + j;       // K index (Cin)
    int col  = t * 16 + (lane & 15);       // N index (Cout)
    WB[tid] = f2bf(W[k * 1024 + krow * 32 + col]);
}

// ---- gather + per-offset GEMM via WMMA: one wave owns a 16x32 output tile ----
__global__ __launch_bounds__(256) void conv_wmma(
    const unsigned short* __restrict__ featsB,   // [N][32] bf16
    const unsigned short* __restrict__ WB,       // [27][2][32][16] bf16 (B layout)
    const int*            __restrict__ nbr,      // [N][27]
    const unsigned char*  __restrict__ mask,     // [N][27] bool
    float*                __restrict__ hraw)     // [N][32] f32 out
{
    __shared__ unsigned int ldsW[KK * 1024 / 2];        // 55296 bytes
    for (int i = threadIdx.x; i < KK * 512; i += 256)
        ldsW[i] = ((const unsigned int*)WB)[i];
    __syncthreads();
    const v4u* bw = (const v4u*)ldsW;

    const int lane = threadIdx.x & 31;
    const int wv   = threadIdx.x >> 5;
    const int r    = lane & 15;
    const int half = lane >> 4;
    const int totalWaves = gridDim.x * 8;

    for (int tile = blockIdx.x * 8 + wv; tile < NTILES; tile += totalWaves) {
        const int base = tile * 16;
        const int row  = base + r;
        v8f acc0 = {};
        v8f acc1 = {};
        const int*           nrow = nbr  + (size_t)row * KK;
        const unsigned char* mrow = mask + (size_t)row * KK;

        for (int k = 0; k < KK; ++k) {
            ABu A, B0, B1;
            v4u z = {0u, 0u, 0u, 0u};
            A.u[0] = z; A.u[1] = z;
            if (mrow[k]) {
                // lane<16: K 0-15 of its row; lane>=16: K 16-31
                const v4u* src = (const v4u*)(featsB + (size_t)nrow[k] * CCH + half * 16);
                A.u[0] = src[0];
                A.u[1] = src[1];
            }
            int b0 = ((k * 2 + 0) * 32 + lane) * 2;
            int b1 = ((k * 2 + 1) * 32 + lane) * 2;
            B0.u[0] = bw[b0];  B0.u[1] = bw[b0 + 1];
            B1.u[0] = bw[b1];  B1.u[1] = bw[b1 + 1];

            acc0 = __builtin_amdgcn_wmma_f32_16x16x32_bf16(false, A.v, false, B0.v,
                                                           (short)0, acc0, false, false);
            acc1 = __builtin_amdgcn_wmma_f32_16x16x32_bf16(false, A.v, false, B1.v,
                                                           (short)0, acc1, false, false);
        }

        // C/D layout: VGPR i -> row M=i (lanes 0-15) / M=i+8 (lanes 16-31), col = lane%16
        float* orow = hraw + (size_t)base * CCH;
        #pragma unroll
        for (int i = 0; i < 8; ++i) {
            int m = i + half * 8;
            orow[m * CCH + r]      = acc0[i];
            orow[m * CCH + 16 + r] = acc1[i];
        }
    }
}

// ---- per-channel sum / sumsq reduction ----
__global__ __launch_bounds__(256) void bn_reduce(const float* __restrict__ x,
                                                 float* __restrict__ sum,
                                                 float* __restrict__ sumsq)
{
    __shared__ float ss[32], sq[32];
    if (threadIdx.x < 32) { ss[threadIdx.x] = 0.f; sq[threadIdx.x] = 0.f; }
    __syncthreads();
    const int c = threadIdx.x & 31;      // stride is multiple of 32 -> channel fixed
    float s = 0.f, q = 0.f;
    const int total = NPTS * CCH;
    for (int i = blockIdx.x * 256 + threadIdx.x; i < total; i += gridDim.x * 256) {
        float v = x[i];
        s += v; q += v * v;
    }
    atomicAdd(&ss[c], s);
    atomicAdd(&sq[c], q);
    __syncthreads();
    if (threadIdx.x < 32) {
        atomicAdd(&sum[threadIdx.x],   ss[threadIdx.x]);
        atomicAdd(&sumsq[threadIdx.x], sq[threadIdx.x]);
    }
}

__global__ void zero_f(float* p, int n) {
    int i = blockIdx.x * blockDim.x + threadIdx.x;
    if (i < n) p[i] = 0.f;
}

__global__ void bn_finalize(const float* __restrict__ sum, const float* __restrict__ sumsq,
                            float* __restrict__ mean, float* __restrict__ rstd)
{
    int c = threadIdx.x;
    if (c < 32) {
        float m = sum[c] * (1.0f / (float)NPTS);
        float v = sumsq[c] * (1.0f / (float)NPTS) - m * m;
        mean[c] = m;
        rstd[c] = rsqrtf(v + EPSV);
    }
}

// ---- BN + ReLU, emit bf16 for the next conv's gather ----
__global__ void bn_relu_bf16(const float* __restrict__ h,
                             const float* __restrict__ mean, const float* __restrict__ rstd,
                             const float* __restrict__ gamma, const float* __restrict__ beta,
                             unsigned short* __restrict__ out, int n)
{
    for (int i = blockIdx.x * blockDim.x + threadIdx.x; i < n; i += gridDim.x * blockDim.x) {
        int c = i & 31;
        float y = (h[i] - mean[c]) * rstd[c] * gamma[c] + beta[c];
        out[i] = f2bf(fmaxf(y, 0.f));
    }
}

// ---- BN2 + residual + ReLU -> final f32 output ----
__global__ void bn_res_relu(const float* __restrict__ h,
                            const float* __restrict__ mean, const float* __restrict__ rstd,
                            const float* __restrict__ gamma, const float* __restrict__ beta,
                            const float* __restrict__ feats,
                            float* __restrict__ out, int n)
{
    for (int i = blockIdx.x * blockDim.x + threadIdx.x; i < n; i += gridDim.x * blockDim.x) {
        int c = i & 31;
        float y = (h[i] - mean[c]) * rstd[c] * gamma[c] + beta[c] + feats[i];
        out[i] = fmaxf(y, 0.f);
    }
}

extern "C" void kernel_launch(void* const* d_in, const int* in_sizes, int n_in,
                              void* d_out, int out_size, void* d_ws, size_t ws_size,
                              hipStream_t stream)
{
    const float*         feats  = (const float*)d_in[0];
    const float*         W1     = (const float*)d_in[1];
    const float*         gamma1 = (const float*)d_in[2];
    const float*         beta1  = (const float*)d_in[3];
    const float*         W2     = (const float*)d_in[4];
    const float*         gamma2 = (const float*)d_in[5];
    const float*         beta2  = (const float*)d_in[6];
    const int*           nbr    = (const int*)d_in[7];
    const unsigned char* mask   = (const unsigned char*)d_in[8];
    float*               out    = (float*)d_out;

    const int NC = NPTS * CCH;

    uint8_t* ws = (uint8_t*)d_ws;
    size_t o = 0;
    unsigned short* featsB = (unsigned short*)(ws + o); o += (size_t)NC * 2;        // 25.6 MB
    unsigned short* h1B    = (unsigned short*)(ws + o); o += (size_t)NC * 2;        // 25.6 MB
    float*          hraw   = (float*)(ws + o);          o += (size_t)NC * 4;        // 51.2 MB (reused)
    unsigned short* WB1    = (unsigned short*)(ws + o); o += (size_t)KK * 1024 * 2; // 55 KB
    unsigned short* WB2    = (unsigned short*)(ws + o); o += (size_t)KK * 1024 * 2; // 55 KB
    float*          stats  = (float*)(ws + o);          o += 8 * 32 * 4;
    float *sum1 = stats,       *sq1 = stats + 32,  *sum2 = stats + 64,  *sq2 = stats + 96;
    float *mu1  = stats + 128, *rs1 = stats + 160, *mu2  = stats + 192, *rs2 = stats + 224;

    // init stats (ws is poisoned by harness)
    zero_f<<<1, 256, 0, stream>>>(stats, 256);

    // precision / layout staging
    cvt_bf16<<<2048, 256, 0, stream>>>(feats, featsB, NC);
    cvt_w<<<(KK * 1024 + 255) / 256, 256, 0, stream>>>(W1, WB1);
    cvt_w<<<(KK * 1024 + 255) / 256, 256, 0, stream>>>(W2, WB2);

    // conv1
    conv_wmma<<<512, 256, 0, stream>>>(featsB, WB1, nbr, mask, hraw);
    bn_reduce<<<1024, 256, 0, stream>>>(hraw, sum1, sq1);
    bn_finalize<<<1, 32, 0, stream>>>(sum1, sq1, mu1, rs1);
    bn_relu_bf16<<<2048, 256, 0, stream>>>(hraw, mu1, rs1, gamma1, beta1, h1B, NC);

    // conv2 (overwrites hraw)
    conv_wmma<<<512, 256, 0, stream>>>(h1B, WB2, nbr, mask, hraw);
    bn_reduce<<<1024, 256, 0, stream>>>(hraw, sum2, sq2);
    bn_finalize<<<1, 32, 0, stream>>>(sum2, sq2, mu2, rs2);

    // BN2 + identity residual + ReLU
    bn_res_relu<<<2048, 256, 0, stream>>>(hraw, mu2, rs2, gamma2, beta2, feats, out, NC);
}
